// AttLayer2_4105988735254
// MI455X (gfx1250) — matrix operand order repaired and checked by reading, not compiled, not run on hardware
//
#include <hip/hip_runtime.h>
#include <hip/hip_bf16.h>

// ---------------------------------------------------------------------------
// AttLayer2 for MI455X (gfx1250, wave32, WMMA + TDM).
// scores = q @ k (9216 x 9216 x 512 ~ 87 GFLOP) dominates -> bf16 WMMA,
// flash-style row-max (scores never materialized). Softmax feeds only an
// equality top-1 mask -> monotonic -> skipped.
// bf16 tensors stored [n][c] row-major. Conv GEMMs: register-staged LDS
// double buffering. Scores GEMM: Tensor Data Mover (tensor_load_to_lds)
// stages Q/K tiles into padded LDS rows, overlapped with WMMA, tracked with
// TENSORcnt (s_wait_tensorcnt).
// ---------------------------------------------------------------------------

typedef __attribute__((ext_vector_type(16))) __bf16 v16bf;
typedef __attribute__((ext_vector_type(8)))  float  v8f;
typedef __attribute__((ext_vector_type(4)))  unsigned int u32x4;
typedef __attribute__((ext_vector_type(8)))  int i32x8;
typedef __attribute__((ext_vector_type(4)))  int i32x4;

#define BSZ   16
#define CSZ   512
#define HWSZ  576            // 24*24
#define NPOS  9216           // BSZ*HWSZ
#define CHWSZ (CSZ*HWSZ)
#define LDP   40             // LDS row pitch in halfs (80B = 64B data + 16B pad)

union PK8 { __bf16 h[8]; uint4 u; };
union PK4 { __bf16 h[4]; uint2 u; };

// ---- WMMA fragment loads from LDS tiles ------------------------------------
// A-operand (16x32 bf16): lane m=L%16, half=L/16; VGPR0-3: k=half*8+{0..7},
// VGPR4-7: k=16+half*8+{0..7}.
__device__ __forceinline__ v16bf load_frag_a(const __bf16* rowp, int half) {
  union { uint4 u[2]; v16bf v; } f;
  f.u[0] = *reinterpret_cast<const uint4*>(rowp + half * 8);
  f.u[1] = *reinterpret_cast<const uint4*>(rowp + 16 + half * 8);
  return f.v;
}
// B-operand (32x16 bf16): lane col n=L%16, khalf=L/16; VGPR0-3: k=khalf*8+{0..7},
// VGPR4-7: k=16+khalf*8+{0..7}.
__device__ __forceinline__ v16bf load_frag_b(const __bf16* rowp, int khalf) {
  union { uint4 u[2]; v16bf v; } f;
  f.u[0] = *reinterpret_cast<const uint4*>(rowp + khalf * 8);
  f.u[1] = *reinterpret_cast<const uint4*>(rowp + 16 + khalf * 8);
  return f.v;
}

// ---- Tensor Data Mover: 2D bf16 tile -> padded LDS rows ---------------------
// D# per CDNA5 ISA: group0 {count=1, lds_addr, global_addr[56:0], type=2},
// group1 {data_size=1(2B), pad_enable, pad_interval=3 (16 DW = 64B rows),
//         pad_amount=3 (4 DW = 16B), dims, tile dims, row stride}.
#if __has_include(<hip/amd_detail/amd_gfx1250_TDM.h>)
#define TDM_CALL(G0, G1) do { i32x4 _z4 = {0,0,0,0}; i32x8 _z8 = {0,0,0,0,0,0,0,0}; \
    __builtin_amdgcn_tensor_load_to_lds((G0), (G1), _z4, _z4, _z8, 0); } while (0)
#else
#define TDM_CALL(G0, G1) do { i32x4 _z4 = {0,0,0,0}; \
    __builtin_amdgcn_tensor_load_to_lds((G0), (G1), _z4, _z4, 0); } while (0)
#endif

__device__ __forceinline__ void tdm_load_tile_bf16(
    unsigned lds_byte_addr, const __bf16* gptr,
    unsigned tile_rows,     // tile_dim1
    unsigned row_elems,     // tile_dim0 = 32 bf16 (64B, multiple of 4B)
    unsigned row_stride)    // elements between rows in memory
{
  unsigned long long ga = (unsigned long long)(size_t)gptr;
  u32x4 g0;
  g0.x = 1u;                                   // count=1, is_restore=0, gather=0
  g0.y = lds_byte_addr;                        // bits 63:32
  g0.z = (unsigned)(ga & 0xffffffffu);         // global_addr[31:0]
  g0.w = (unsigned)((ga >> 32) & 0x01ffffffu)  // global_addr[56:32]
       | (2u << 30);                           // type=2 ("image")
  i32x8 g1;
  g1[0] = (int)((1u << 16)        // data_size = 2 bytes
              | (1u << 20)        // pad_enable
              | (3u << 22)        // pad_interval: 16 DWORDs (64B) between pads
              | (3u << 25));      // pad_amount:   4 DWORDs (16B) pad
  g1[1] = (int)((row_elems & 0xffffu) << 16);            // tensor_dim0[15:0]
  g1[2] = (int)((row_elems >> 16) | ((tile_rows & 0xffffu) << 16)); // d0 hi | tensor_dim1 lo
  g1[3] = (int)((tile_rows >> 16) | ((row_elems & 0xffffu) << 16)); // d1 hi | tile_dim0
  g1[4] = (int)(tile_rows & 0xffffu);                    // tile_dim1 (tile_dim2=0)
  g1[5] = (int)row_stride;                               // tensor_dim0_stride[31:0]
  g1[6] = 0;                                             // stride hi, dim1_stride lo
  g1[7] = 0;
  TDM_CALL(g0, g1);
}

// ---------------------------------------------------------------------------
// 1x1-conv GEMM:  Y[o,n] = sum_c W[o,c] * X[c,n] + bias[o] (+ residual)
// Block tile 128(o) x 64(n); 8 waves = 4(o-sub32) x 2(n-sub32); 4 WMMA/K-step.
// K step 32, double-buffered LDS, register-staged copies.
// mode 0: X from f32 NCHW input; writes f32 NCHW (with residual) + bf16 [n][c]
// mode 1: X from bf16 [n][c]; writes bf16 [n][c]
// ---------------------------------------------------------------------------
__global__ __launch_bounds__(256)
void conv1x1_wmma(const float* __restrict__ xin,
                  const __bf16* __restrict__ xbf,
                  const float* __restrict__ Wm,     // (O=C, K=C) row-major f32
                  const float* __restrict__ bias,
                  float* __restrict__ out_f32,
                  __bf16* __restrict__ out_bf,      // [n][c]
                  int mode) {
  __shared__ __bf16 Wt[2][128][LDP];
  __shared__ __bf16 Xt[2][64][LDP];
  __shared__ float  sBias[128];

  const int tid  = threadIdx.x;
  const int lane = tid & 31, wave = tid >> 5;
  const int osub = wave & 3;          // 4 x 32-o groups
  const int nsub = wave >> 2;         // 2 x 32-n groups
  const int n0   = blockIdx.x * 64;   // 144 tiles, each inside one image
  const int o0   = blockIdx.y * 128;  // 4 tiles
  const int b    = n0 / HWSZ;
  const int hwb  = n0 - b * HWSZ;
  const int half = lane >> 4;
  const int l16  = lane & 15;

  float4 wreg[4];                 // 128x32 f32 weights: 4 float4 / thread
  float4 xregf[2];                // mode 0: 64x32 f32 acts (transpose)
  uint4  xregb;                   // mode 1: 64x32 bf16 acts: 1 uint4 / thread

  const int w_o   = tid >> 2;
  const int x_row = tid >> 2;           // bf16 acts: 64 rows, 4 thr/row
  const int x_c8  = (tid & 3) * 8;

  if (tid < 128) sBias[tid] = bias[o0 + tid];

#define LOAD_W(K0)                                                            \
  _Pragma("unroll")                                                           \
  for (int it = 0; it < 4; ++it) {                                            \
    int idx = tid + it * 256;     /* 1024 float4 over 128x32 */               \
    int c4 = (idx & 7) * 4, o = idx >> 3;                                     \
    wreg[it] = *reinterpret_cast<const float4*>(&Wm[(size_t)(o0 + o) * CSZ + (K0) + c4]); \
  }
#define STORE_W(BUF)                                                          \
  _Pragma("unroll")                                                           \
  for (int it = 0; it < 4; ++it) {                                            \
    int idx = tid + it * 256;                                                 \
    int c4 = (idx & 7) * 4, o = idx >> 3;                                     \
    PK4 p; p.h[0] = (__bf16)wreg[it].x; p.h[1] = (__bf16)wreg[it].y;          \
    p.h[2] = (__bf16)wreg[it].z; p.h[3] = (__bf16)wreg[it].w;                 \
    *reinterpret_cast<uint2*>(&Wt[BUF][o][c4]) = p.u;                         \
  }
#define LOAD_X(K0)                                                            \
  if (mode == 0) {                                                            \
    _Pragma("unroll")                                                         \
    for (int it = 0; it < 2; ++it) {                                          \
      int idx = tid + it * 256;   /* 512 float4 over 64n x 32c (transpose) */ \
      int i4 = (idx & 15) * 4, c = idx >> 4;                                  \
      xregf[it] = *reinterpret_cast<const float4*>(                           \
          &xin[(size_t)b * CHWSZ + (size_t)((K0) + c) * HWSZ + hwb + i4]);    \
    }                                                                         \
  } else {                                                                    \
    xregb = *reinterpret_cast<const uint4*>(                                  \
        &xbf[(size_t)(n0 + x_row) * CSZ + (K0) + x_c8]);                      \
  }
#define STORE_X(BUF)                                                          \
  if (mode == 0) {                                                            \
    _Pragma("unroll")                                                         \
    for (int it = 0; it < 2; ++it) {                                          \
      int idx = tid + it * 256;                                               \
      int i4 = (idx & 15) * 4, c = idx >> 4;                                  \
      float fv[4] = {xregf[it].x, xregf[it].y, xregf[it].z, xregf[it].w};     \
      _Pragma("unroll")                                                       \
      for (int j = 0; j < 4; ++j) Xt[BUF][i4 + j][c] = (__bf16)fv[j];         \
    }                                                                         \
  } else {                                                                    \
    *reinterpret_cast<uint4*>(&Xt[BUF][x_row][x_c8]) = xregb;                 \
  }

  v8f acc00 = {}, acc01 = {}, acc10 = {}, acc11 = {};

  LOAD_W(0); LOAD_X(0);
  STORE_W(0); STORE_X(0);

  for (int kt = 0; kt < CSZ / 32; ++kt) {
    __syncthreads();
    const int cb = kt & 1, nxt = kt + 1;
    if (nxt < CSZ / 32) {
      __builtin_prefetch(&Wm[(size_t)(o0 + w_o) * CSZ + nxt * 32 + 32], 0, 1);
      LOAD_W(nxt * 32); LOAD_X(nxt * 32);
    }
    v16bf a0 = load_frag_a(&Wt[cb][osub * 32 + l16][0], half);
    v16bf a1 = load_frag_a(&Wt[cb][osub * 32 + 16 + l16][0], half);
    v16bf bf0 = load_frag_b(&Xt[cb][nsub * 32 + l16][0], half);
    v16bf bf1 = load_frag_b(&Xt[cb][nsub * 32 + 16 + l16][0], half);
    acc00 = __builtin_amdgcn_wmma_f32_16x16x32_bf16(false, a0, false, bf0, (short)0, acc00, false, false);
    acc01 = __builtin_amdgcn_wmma_f32_16x16x32_bf16(false, a0, false, bf1, (short)0, acc01, false, false);
    acc10 = __builtin_amdgcn_wmma_f32_16x16x32_bf16(false, a1, false, bf0, (short)0, acc10, false, false);
    acc11 = __builtin_amdgcn_wmma_f32_16x16x32_bf16(false, a1, false, bf1, (short)0, acc11, false, false);
    if (nxt < CSZ / 32) { STORE_W(nxt & 1); STORE_X(nxt & 1); }
  }

  // D layout: lane L, reg r -> M(o) = r + 8*(L/16), N(n) = L%16
  #pragma unroll
  for (int aA = 0; aA < 2; ++aA) {
    #pragma unroll
    for (int bB = 0; bB < 2; ++bB) {
      v8f A = aA ? (bB ? acc11 : acc10) : (bB ? acc01 : acc00);
      const int n = n0 + nsub * 32 + bB * 16 + l16;
      const int obase = o0 + osub * 32 + aA * 16 + 8 * half;
      PK8 pk;
      #pragma unroll
      for (int r = 0; r < 8; ++r) {
        const int o = obase + r;
        float v = A[r] + sBias[o - o0];
        if (mode == 0) {
          size_t gi = (size_t)b * CHWSZ + (size_t)o * HWSZ + (n - b * HWSZ);
          v += xin[gi];              // residual; coalesced along l16
          out_f32[gi] = v;
        }
        pk.h[r] = (__bf16)v;
      }
      *reinterpret_cast<uint4*>(&out_bf[(size_t)n * CSZ + obase]) = pk.u; // b128
    }
  }
}

// ---------------------------------------------------------------------------
// scores row-max:  smax[n][b] = max_m sum_c q[n,c] * k[b,c,m]
// Block = 128-query tile x one image; 9 key tiles of 64; K=512 bf16 WMMA.
// 8 waves = 4(q-sub32) x 2(key-sub32), 2x2 fragments, 4 WMMA / K-step.
// Q/K tiles staged by the Tensor Data Mover into padded LDS rows, double
// buffered; wave 0 issues the DMA, s_wait_tensorcnt + barrier synchronize.
// ---------------------------------------------------------------------------
__global__ __launch_bounds__(256)
void scores_rowmax(const __bf16* __restrict__ qbf,  // [n][c]
                   const __bf16* __restrict__ kbf,  // [n][c]
                   float* __restrict__ smax) {      // [n][b]
  __shared__ __bf16 Qt[2][128][LDP];
  __shared__ __bf16 Kt[2][64][LDP];
  __shared__ float  red[128][2];

  const int tid  = threadIdx.x;
  const int lane = tid & 31, wave = tid >> 5;
  const int nsub = wave & 3;       // 4 x 32 query rows
  const int msub = wave >> 2;      // 2 x 32 key cols
  const int n0   = blockIdx.x * 128;
  const int bimg = blockIdx.y;
  const int half = lane >> 4, l16 = lane & 15;

  const unsigned qt_lds[2] = { (unsigned)(size_t)&Qt[0][0][0],
                               (unsigned)(size_t)&Qt[1][0][0] };
  const unsigned kt_lds[2] = { (unsigned)(size_t)&Kt[0][0][0],
                               (unsigned)(size_t)&Kt[1][0][0] };

  float rmax[2][2][8];
  #pragma unroll
  for (int aA = 0; aA < 2; ++aA)
    #pragma unroll
    for (int bB = 0; bB < 2; ++bB)
      #pragma unroll
      for (int r = 0; r < 8; ++r) rmax[aA][bB][r] = -__builtin_inff();

  for (int m0 = 0; m0 < HWSZ; m0 += 64) {
    v8f acc00 = {}, acc01 = {}, acc10 = {}, acc11 = {};
    if (wave == 0) {                       // DMA tile 0 (TENSORcnt += 2)
      tdm_load_tile_bf16(qt_lds[0], &qbf[(size_t)n0 * CSZ], 128, 32, CSZ);
      tdm_load_tile_bf16(kt_lds[0], &kbf[(size_t)(bimg * HWSZ + m0) * CSZ], 64, 32, CSZ);
      __builtin_amdgcn_s_wait_tensorcnt(0);
    }
    for (int kt = 0; kt < CSZ / 32; ++kt) {
      __syncthreads();                     // buf[kt&1] staged; buf[nxt&1] free
      const int cb = kt & 1, nxt = kt + 1;
      if (wave == 0 && nxt < CSZ / 32) {   // DMA next tile while computing
        tdm_load_tile_bf16(qt_lds[nxt & 1], &qbf[(size_t)n0 * CSZ + nxt * 32], 128, 32, CSZ);
        tdm_load_tile_bf16(kt_lds[nxt & 1], &kbf[(size_t)(bimg * HWSZ + m0) * CSZ + nxt * 32], 64, 32, CSZ);
      }
      v16bf a0 = load_frag_a(&Qt[cb][nsub * 32 + l16][0], half);
      v16bf a1 = load_frag_a(&Qt[cb][nsub * 32 + 16 + l16][0], half);
      v16bf b0 = load_frag_b(&Kt[cb][msub * 32 + l16][0], half);
      v16bf b1 = load_frag_b(&Kt[cb][msub * 32 + 16 + l16][0], half);
      acc00 = __builtin_amdgcn_wmma_f32_16x16x32_bf16(false, a0, false, b0, (short)0, acc00, false, false);
      acc01 = __builtin_amdgcn_wmma_f32_16x16x32_bf16(false, a0, false, b1, (short)0, acc01, false, false);
      acc10 = __builtin_amdgcn_wmma_f32_16x16x32_bf16(false, a1, false, b0, (short)0, acc10, false, false);
      acc11 = __builtin_amdgcn_wmma_f32_16x16x32_bf16(false, a1, false, b1, (short)0, acc11, false, false);
      if (wave == 0 && nxt < CSZ / 32) __builtin_amdgcn_s_wait_tensorcnt(0);
    }
    #pragma unroll
    for (int r = 0; r < 8; ++r) {
      rmax[0][0][r] = fmaxf(rmax[0][0][r], acc00[r]);
      rmax[0][1][r] = fmaxf(rmax[0][1][r], acc01[r]);
      rmax[1][0][r] = fmaxf(rmax[1][0][r], acc10[r]);
      rmax[1][1][r] = fmaxf(rmax[1][1][r], acc11[r]);
    }
    __syncthreads();                       // all waves done before next m0 DMA
  }

  // reduce over key cols: lane%16 carries m; combine bB, then xor within halves
  #pragma unroll
  for (int aA = 0; aA < 2; ++aA) {
    #pragma unroll
    for (int r = 0; r < 8; ++r) {
      float v = fmaxf(rmax[aA][0][r], rmax[aA][1][r]);
      #pragma unroll
      for (int off = 1; off < 16; off <<= 1) v = fmaxf(v, __shfl_xor(v, off, 32));
      if (l16 == 0) red[nsub * 32 + aA * 16 + 8 * half + r][msub] = v;
    }
  }
  __syncthreads();
  if (tid < 128)
    smax[(size_t)(n0 + tid) * BSZ + bimg] = fmaxf(red[tid][0], red[tid][1]);
}

// ---- small VALU kernels -----------------------------------------------------

__global__ void xw_mean(const float* __restrict__ smax, float* __restrict__ xw) {
  int n = blockIdx.x * blockDim.x + threadIdx.x;
  if (n >= NPOS) return;
  float s = 0.f;
  #pragma unroll
  for (int b = 0; b < BSZ; ++b) s += smax[(size_t)n * BSZ + b];
  xw[n] = s * (0.044194173824159216f / 16.0f);  // scale=1/sqrt(512), mean of 16
}

__global__ __launch_bounds__(256)
void mask_top1(const float* __restrict__ xw, float* __restrict__ mask_out) {
  __shared__ float sred[256];
  int b = blockIdx.x, tid = threadIdx.x;
  float mx = -__builtin_inff();
  for (int hw = tid; hw < HWSZ; hw += 256) mx = fmaxf(mx, xw[b * HWSZ + hw]);
  sred[tid] = mx; __syncthreads();
  for (int s = 128; s > 0; s >>= 1) {
    if (tid < s) sred[tid] = fmaxf(sred[tid], sred[tid + s]);
    __syncthreads();
  }
  float rowmax = sred[0];
  for (int hw = tid; hw < HWSZ; hw += 256)
    mask_out[b * HWSZ + hw] = (xw[b * HWSZ + hw] == rowmax) ? 1.0f : 0.0f;
}

__global__ void invnorm_k(const float* __restrict__ x5, float* __restrict__ invn) {
  int n = blockIdx.x * blockDim.x + threadIdx.x;
  if (n >= NPOS) return;
  int b = n / HWSZ, hw = n - b * HWSZ;
  const float* p = x5 + (size_t)b * CHWSZ + hw;
  float s = 0.f;
  for (int c = 0; c < CSZ; ++c) { float v = p[(size_t)c * HWSZ]; s += v * v; }
  invn[n] = 1.0f / fmaxf(sqrtf(s), 1e-12f);
}

__global__ __launch_bounds__(256)
void seeds_k(const float* __restrict__ x5, const float* __restrict__ mask,
             const float* __restrict__ invn, float* __restrict__ seeds) {
  int b = blockIdx.x, tid = threadIdx.x;
  int c0 = tid, c1 = tid + 256;
  float a0 = 0.f, a1 = 0.f;
  const float* xb = x5 + (size_t)b * CHWSZ;
  for (int hw = 0; hw < HWSZ; ++hw) {
    float mv = mask[b * HWSZ + hw];
    if (mv != 0.0f) {                        // ~1 hot location per image
      float inr = invn[b * HWSZ + hw] * mv;
      a0 += xb[(size_t)c0 * HWSZ + hw] * inr;
      a1 += xb[(size_t)c1 * HWSZ + hw] * inr;
    }
  }
  seeds[b * CSZ + c0] = a0;
  seeds[b * CSZ + c1] = a1;
}

__global__ __launch_bounds__(256)
void corr_k(const float* __restrict__ x5, const float* __restrict__ invn,
            const float* __restrict__ seeds, float* __restrict__ corr) {
  __shared__ float sld[BSZ * CSZ];   // 32 KB
  int tid = threadIdx.x;
  for (int i = tid; i < BSZ * CSZ; i += 256) sld[i] = seeds[i];
  __syncthreads();
  int n = blockIdx.x * 256 + tid;    // grid = 36 -> exact
  int b = n / HWSZ, hw = n - b * HWSZ;
  float inr = invn[n];
  const float* xb = x5 + (size_t)b * CHWSZ + hw;
  float acc[BSZ];
  #pragma unroll
  for (int s = 0; s < BSZ; ++s) acc[s] = 0.f;
  for (int c = 0; c < CSZ; ++c) {
    float v = xb[(size_t)c * HWSZ] * inr;
    #pragma unroll
    for (int s = 0; s < BSZ; ++s) acc[s] += v * sld[s * CSZ + c];
  }
  float m = 0.f;
  #pragma unroll
  for (int s = 0; s < BSZ; ++s) m += fmaxf(acc[s], 0.f);
  corr[n] = m * (1.0f / BSZ);
}

__global__ __launch_bounds__(256)
void cormap_k(const float* __restrict__ corr, float* __restrict__ cormap) {
  __shared__ float smn[256], smx[256];
  int b = blockIdx.x, tid = threadIdx.x;
  float mn = __builtin_inff(), mx = -__builtin_inff();
  for (int hw = tid; hw < HWSZ; hw += 256) {
    float v = corr[b * HWSZ + hw];
    mn = fminf(mn, v); mx = fmaxf(mx, v);
  }
  smn[tid] = mn; smx[tid] = mx; __syncthreads();
  for (int s = 128; s > 0; s >>= 1) {
    if (tid < s) { smn[tid] = fminf(smn[tid], smn[tid + s]); smx[tid] = fmaxf(smx[tid], smx[tid + s]); }
    __syncthreads();
  }
  float lo = smn[0], inv = 1.0f / (smx[0] - smn[0] + 1e-12f);
  for (int hw = tid; hw < HWSZ; hw += 256)
    cormap[b * HWSZ + hw] = (corr[b * HWSZ + hw] - lo) * inv;
}

__global__ __launch_bounds__(256)
void proto_k(const float* __restrict__ x5, const float* __restrict__ cormap,
             float* __restrict__ proto) {
  __shared__ float sred[256];
  int c = blockIdx.x, tid = threadIdx.x;
  float s = 0.f;
  for (int n = tid; n < NPOS; n += 256) {
    int b = n / HWSZ, hw = n - b * HWSZ;
    s += x5[(size_t)b * CHWSZ + (size_t)c * HWSZ + hw] * cormap[n];
  }
  sred[tid] = s; __syncthreads();
  for (int st = 128; st > 0; st >>= 1) {
    if (tid < st) sred[tid] += sred[tid + st];
    __syncthreads();
  }
  if (tid == 0) proto[c] = sred[0] * (1.0f / NPOS);
}

__global__ void final_k(const float* __restrict__ x5, const float* __restrict__ proto,
                        const float* __restrict__ cormap, float* __restrict__ out2) {
  int idx = blockIdx.x * blockDim.x + threadIdx.x;   // grid 18432*256 exact
  int hw = idx % HWSZ;
  int c  = (idx / HWSZ) & (CSZ - 1);
  int b  = idx / CHWSZ;
  out2[idx] = x5[idx] * (proto[c] + cormap[b * HWSZ + hw]);
}

// ---------------------------------------------------------------------------
extern "C" void kernel_launch(void* const* d_in, const int* in_sizes, int n_in,
                              void* d_out, int out_size, void* d_ws, size_t ws_size,
                              hipStream_t stream) {
  (void)in_sizes; (void)n_in; (void)out_size; (void)ws_size;
  const float* x5 = (const float*)d_in[0];
  const float* cw = (const float*)d_in[1];
  const float* cb = (const float*)d_in[2];
  const float* qw = (const float*)d_in[3];
  const float* qb = (const float*)d_in[4];
  const float* kw = (const float*)d_in[5];
  const float* kb = (const float*)d_in[6];

  float* out_x5   = (float*)d_out;                 // (B,C,H,W)    4,718,592
  float* out_pr   = out_x5 + (size_t)CHWSZ * BSZ;  // proto1       512
  float* out_mix  = out_pr + CSZ;                  // x5*p + x51   4,718,592
  float* out_mask = out_mix + (size_t)CHWSZ * BSZ; // mask         9,216

  // scratch carve-out (~29 MB needed)
  char* ws = (char*)d_ws; size_t off = 0;
  auto carve = [&](size_t bytes) -> void* {
    void* p = ws + off; off = (off + bytes + 255) & ~(size_t)255; return p;
  };
  __bf16* x5bf = (__bf16*)carve((size_t)NPOS * CSZ * 2);   // [n][c]
  __bf16* qbf  = (__bf16*)carve((size_t)NPOS * CSZ * 2);   // [n][c]
  __bf16* kbf  = (__bf16*)carve((size_t)NPOS * CSZ * 2);   // [n][c]
  float* smax   = (float*)carve((size_t)NPOS * BSZ * 4);
  float* xw     = (float*)carve((size_t)NPOS * 4);
  float* invn   = (float*)carve((size_t)NPOS * 4);
  float* seeds  = (float*)carve((size_t)BSZ * CSZ * 4);
  float* corr   = (float*)carve((size_t)NPOS * 4);
  float* cormap = (float*)carve((size_t)NPOS * 4);

  dim3 blk(256);
  // 1) x5' = conv + bias + residual  (f32 NCHW out + bf16 [n][c] scratch)
  conv1x1_wmma<<<dim3(144, 4), blk, 0, stream>>>(x5, nullptr, cw, cb, out_x5, x5bf, 0);
  // 2) q, k projections (bf16 [n][c])
  conv1x1_wmma<<<dim3(144, 4), blk, 0, stream>>>(nullptr, x5bf, qw, qb, nullptr, qbf, 1);
  conv1x1_wmma<<<dim3(144, 4), blk, 0, stream>>>(nullptr, x5bf, kw, kb, nullptr, kbf, 1);
  // 3) flash-style score row-max per (128 queries, image), TDM-staged tiles
  scores_rowmax<<<dim3(72, BSZ), blk, 0, stream>>>(qbf, kbf, smax);
  // 4) mean over images (scale folded), top-1 equality mask (softmax skipped)
  xw_mean<<<36, 256, 0, stream>>>(smax, xw);
  mask_top1<<<BSZ, 256, 0, stream>>>(xw, out_mask);
  // 5) channel-L2 norms, seeds, seeded correlation, min-max map
  invnorm_k<<<36, 256, 0, stream>>>(out_x5, invn);
  seeds_k<<<BSZ, 256, 0, stream>>>(out_x5, out_mask, invn, seeds);
  corr_k<<<36, 256, 0, stream>>>(out_x5, invn, seeds, corr);
  cormap_k<<<BSZ, 256, 0, stream>>>(corr, cormap);
  // 6) proto1 and fused final output
  proto_k<<<CSZ, 256, 0, stream>>>(out_x5, cormap, out_pr);
  final_k<<<18432, 256, 0, stream>>>(out_x5, out_pr, cormap, out_mix);
}
